// Network2_39307540693900
// MI455X (gfx1250) — matrix operand order repaired and checked by reading, not compiled, not run on hardware
//
#include <hip/hip_runtime.h>
#include <hip/hip_bf16.h>
#include <stdint.h>

// ---------------------------------------------------------------------------
// SLAYER SNN forward for gfx1250 (MI455X), wave32 + WMMA f16 + TDM/LDS staging.
// Layers: conv1 -> psp -> spike -> pool -> psp -> spike -> conv2(WMMA) ->
//         psp -> spike -> fc1(WMMA) -> psp -> spike -> fc2(WMMA) -> psp ->
//         spike -> fc3(WMMA) -> psp -> spike -> out [10][2048] f32
// Temporal kernels k*e^(1-k/tau) implemented as exact 2-pole IIRs.
// ---------------------------------------------------------------------------

typedef __attribute__((ext_vector_type(16))) _Float16 v16h;
typedef __attribute__((ext_vector_type(8)))  _Float16 v8h;
typedef __attribute__((ext_vector_type(8)))  float    v8f;
typedef unsigned int uint32x4 __attribute__((ext_vector_type(4)));
typedef int          int32x8  __attribute__((ext_vector_type(8)));
typedef int          int32x4  __attribute__((ext_vector_type(4)));

#if defined(__has_builtin)
# if __has_builtin(__builtin_amdgcn_tensor_load_to_lds) && \
     __has_builtin(__builtin_amdgcn_s_wait_tensorcnt)
#  define USE_TDM 1
# endif
#endif
#ifndef USE_TDM
# define USE_TDM 0
#endif

#define T_BINS 2048

// PSP (SRM) kernel eps(k) = (k/10) e^{1-k/10}: pole b = e^{-0.1}
#define PSP_2B  1.8096748360719190f   // 2*e^{-0.1}
#define PSP_B2  0.8187307530779818f   // e^{-0.2}
#define PSP_H1  0.2459603111156950f   // eps(1) = 0.1*e^{0.9}
// Refractory kernel ref(j) = -2*theta*j*e^{1-j}: pole a = e^{-1}
#define REF_2A  0.7357588823428847f   // 2*e^{-1}
#define REF_A2  0.1353352832366127f   // e^{-2}
#define REF_G1 -2.0f                  // ref(1)

struct ScanState {
    float y1, y2, xp, r1, r2, sp;
    __device__ inline void init() { y1 = y2 = xp = r1 = r2 = sp = 0.0f; }
    // x = linear-layer output at time t; returns spike s[t]
    __device__ inline float step(float x) {
        float y = PSP_2B * y1 - PSP_B2 * y2 + PSP_H1 * xp;  // psp[t]
        float r = REF_2A * r1 - REF_A2 * r2 + REF_G1 * sp;  // refractory[t]
        float u = y + r;                                     // membrane
        float s = (u >= 1.0f) ? 1.0f : 0.0f;                 // theta = 1
        y2 = y1; y1 = y; xp = x;
        r2 = r1; r1 = r; sp = s;
        return s;
    }
};

// --------------------------- weight pad/convert ----------------------------
__global__ void pad_convert_f16(const float* __restrict__ src,
                                _Float16* __restrict__ dst,
                                int Msrc, int Ksrc, int Mdst, int Kdst) {
    int i = blockIdx.x * blockDim.x + threadIdx.x;
    int total = Mdst * Kdst;
    if (i >= total) return;
    int m = i / Kdst, k = i % Kdst;
    float v = (m < Msrc && k < Ksrc) ? src[m * Ksrc + k] : 0.0f;
    dst[i] = (_Float16)v;
}

// ------------------------- layer 1: conv1 + scan ---------------------------
// x [3][32][32], Wc1 [6][3][5][5]; conv output is time-constant.
// s1 layout: [neuron(4704)][T] f16.
__global__ void conv1_psp_spike(const float* __restrict__ x,
                                const float* __restrict__ Wc1,
                                _Float16* __restrict__ s1) {
    int n = blockIdx.x * blockDim.x + threadIdx.x;
    if (n >= 6 * 28 * 28) return;
    int co = n / 784, rem = n % 784;
    int y = rem / 28, xx = rem % 28;
    float c = 0.0f;
    #pragma unroll
    for (int ci = 0; ci < 3; ++ci)
        #pragma unroll
        for (int ky = 0; ky < 5; ++ky)
            #pragma unroll
            for (int kx = 0; kx < 5; ++kx)
                c += x[ci * 1024 + (y + ky) * 32 + (xx + kx)] *
                     Wc1[co * 75 + ci * 25 + ky * 5 + kx];
    ScanState st; st.init();
    _Float16* outp = s1 + (size_t)n * T_BINS;
    for (int t = 0; t < T_BINS; ++t)
        outp[t] = (_Float16)st.step(c);
}

// ------------------------- layer 2: pool + scan ----------------------------
// pool = 1.1 * sum(2x2).  s2 layout: [T][1184] f16 (features 0..1175 valid).
__global__ void pool_psp_spike(const _Float16* __restrict__ s1,
                               _Float16* __restrict__ s2) {
    int f = blockIdx.x * blockDim.x + threadIdx.x;
    if (f >= 6 * 14 * 14) return;
    int ci = f / 196, rem = f % 196;
    int y = rem / 14, xx = rem % 14;
    const _Float16* r0 = s1 + (size_t)(ci * 784 + (2 * y) * 28 + 2 * xx) * T_BINS;
    const _Float16* r1 = r0 + T_BINS;
    const _Float16* r2 = s1 + (size_t)(ci * 784 + (2 * y + 1) * 28 + 2 * xx) * T_BINS;
    const _Float16* r3 = r2 + T_BINS;
    ScanState st; st.init();
    for (int t = 0; t < T_BINS; ++t) {
        float xin = 1.1f * ((float)r0[t] + (float)r1[t] + (float)r2[t] + (float)r3[t]);
        s2[(size_t)t * 1184 + f] = (_Float16)st.step(xin);
    }
}

// ------------------------- conv2 im2col (chunked) --------------------------
// Bim [20 positions][T][160] f16: column (pos,t), K = (ci,ky,kx) padded to 160.
__global__ void im2col_conv2(const _Float16* __restrict__ s2,
                             _Float16* __restrict__ Bim, int chunkStart) {
    int idx = blockIdx.x * blockDim.x + threadIdx.x;
    if (idx >= 20 * 2048 * 160) return;
    int k = idx % 160;
    int t = (idx / 160) % 2048;
    int c = idx / (160 * 2048);
    _Float16 v = (_Float16)0.0f;
    if (k < 150) {
        int pos = chunkStart + c;
        int y = pos / 10, xx = pos % 10;
        int ci = k / 25, r = k % 25, ky = r / 5, kx = r % 5;
        v = s2[(size_t)t * 1184 + ci * 196 + (y + ky) * 14 + (xx + kx)];
    }
    Bim[idx] = v;
}

// --------------------------- generic WMMA GEMM -----------------------------
// C[M][ldc] f32 = A[Mt*16][K] f16 (row major) x B^T, where B is stored as
// Bt[N][K] f16 (each column's K-run contiguous).  One wave = one 16x16 tile.
// Mapping: 8 waves per block share one M tile (Nt % 8 == 0), so the A tile
// for each K step is staged ONCE per block into LDS (via the Tensor Data
// Mover when available) and consumed by all 8 waves through ds_load_b128.
// B columns are wave-private and stream directly via global_load_b128.
__global__ void wmma_gemm_f16(const _Float16* __restrict__ A,
                              const _Float16* __restrict__ Bt,
                              float* __restrict__ C,
                              int Mt, int Nt, int K, int ldc, int Mvalid) {
    __shared__ _Float16 atile[16 * 32];   // one K-step A tile (1 KB)

    int wv = threadIdx.x >> 5;
    int wid = blockIdx.x * 8 + wv;
    if (wid >= Mt * Nt) return;           // never triggers (grids are exact)
    int lane = threadIdx.x & 31;
    int half = lane >> 4;                 // 0 or 1
    int l16  = lane & 15;
    int m0 = (wid / Nt) * 16;             // shared by all 8 waves in block
    int n0 = (wid % Nt) * 16;

    // B lane base: column n0+l16; K chunks half*8 and half*8+16.
    const _Float16* bcol = Bt + (size_t)(n0 + l16) * K + half * 8;
    // LDS A fragment for this lane: row l16, K chunks half*8 and half*8+16.
    const v8h* afrag = (const v8h*)(atile + l16 * 32 + half * 8);

#if USE_TDM
    uint32_t ldsaddr = (uint32_t)(size_t)(void*)atile;
    uint64_t gabase  = (uint64_t)(size_t)(A + (size_t)m0 * K);
#endif

    v8f acc = {};
    int ksteps = K >> 5;
    for (int kk = 0; kk < ksteps; ++kk) {
        __syncthreads();                  // previous A tile fully consumed
#if USE_TDM
        if (wv == 0) {
            // D# group0: count=1 | lds_addr | global_addr | type=2
            uint64_t ga = gabase + (uint64_t)kk * 64;   // 32 halfs per step
            uint32x4 g0;
            g0[0] = 1u;                                 // count=1, user desc
            g0[1] = ldsaddr;                            // LDS byte address
            g0[2] = (uint32_t)ga;                       // global_addr[31:0]
            g0[3] = (uint32_t)(ga >> 32) | 0x80000000u; // [56:32] | type=2
            // D# group1: 2D tile 32x16 elem of 2B, row stride K elements
            int32x8 g1;
            g1[0] = 0x00010000;                         // data_size=1 (2B)
            g1[1] = (int)((K & 0xFFFF) << 16);          // tensor_dim0 lo
            g1[2] = (int)(((unsigned)K >> 16) & 0xFFFF) | (16 << 16); // dim0 hi | tensor_dim1=16
            g1[3] = (int)(32u << 16);                   // tile_dim0 = 32
            g1[4] = 16;                                 // tile_dim1 = 16
            g1[5] = (int)K;                             // tensor_dim0_stride
            g1[6] = 0;
            g1[7] = 0;
            int32x4 z4 = {};
            int32x8 z8 = {};
            __builtin_amdgcn_tensor_load_to_lds(g0, g1, z4, z4, z8, 0);
            __builtin_amdgcn_s_wait_tensorcnt(0);
        }
#else
        {   // cooperative fallback: 256 threads x 4B = 1 KB tile
            int d = threadIdx.x;
            int r = d >> 4, w = d & 15;
            ((uint32_t*)atile)[d] =
                ((const uint32_t*)(A + (size_t)(m0 + r) * K + kk * 32))[w];
        }
#endif
        __syncthreads();                  // A tile visible to all waves

        const v8h* bp = (const v8h*)(bcol + kk * 32);
        __builtin_prefetch((const void*)(bcol + (kk + 1) * 32), 0, 0);
        v8h alo = afrag[0];               // ds_load_b128
        v8h ahi = afrag[2];               // +16 halves
        v8h blo = bp[0];
        v8h bhi = bp[2];
        v16h a, b;
        #pragma unroll
        for (int i = 0; i < 8; ++i) {
            a[i] = alo[i]; a[i + 8] = ahi[i];
            b[i] = blo[i]; b[i + 8] = bhi[i];
        }
        acc = __builtin_amdgcn_wmma_f32_16x16x32_f16(
            /*neg_a=*/false, a, /*neg_b=*/false, b,
            /*c_mod=*/(short)0, acc, /*reuse_a=*/false, /*reuse_b=*/false);
    }

    float* cbase = C + (size_t)n0 + l16;
    #pragma unroll
    for (int r = 0; r < 8; ++r) {
        int m = m0 + r + half * 8;
        if (m < Mvalid) cbase[(size_t)m * ldc] = acc[r];
    }
}

// ------------------- psp+spike scan: rows f32 -> cols f16 ------------------
// P [rows][T] f32 membranes; S [T][Kp] f16 spikes (features >= valid -> 0).
__global__ void psp_spike_r2c(const float* __restrict__ P,
                              _Float16* __restrict__ S, int valid, int Kp) {
    int i = blockIdx.x * blockDim.x + threadIdx.x;
    if (i >= Kp) return;
    if (i >= valid) {
        for (int t = 0; t < T_BINS; ++t) S[(size_t)t * Kp + i] = (_Float16)0.0f;
        return;
    }
    const float* p = P + (size_t)i * T_BINS;
    ScanState st; st.init();
    for (int t = 0; t < T_BINS; ++t)
        S[(size_t)t * Kp + i] = (_Float16)st.step(p[t]);
}

// ----------------------------- final layer scan ----------------------------
__global__ void psp_spike_final(const float* __restrict__ P,
                                float* __restrict__ out) {
    int i = threadIdx.x;
    if (i >= 10) return;
    const float* p = P + (size_t)i * T_BINS;
    float* o = out + (size_t)i * T_BINS;
    ScanState st; st.init();
    for (int t = 0; t < T_BINS; ++t)
        o[t] = st.step(p[t]);
}

// ---------------------------------------------------------------------------
extern "C" void kernel_launch(void* const* d_in, const int* in_sizes, int n_in,
                              void* d_out, int out_size, void* d_ws, size_t ws_size,
                              hipStream_t stream) {
    const float* x   = (const float*)d_in[0];   // [1,3,32,32]
    const float* Wc1 = (const float*)d_in[1];   // [6,3,5,5]
    const float* Wc2 = (const float*)d_in[2];   // [16,6,5,5]
    const float* Wf1 = (const float*)d_in[3];   // [300,1600]
    const float* Wf2 = (const float*)d_in[4];   // [150,300]
    const float* Wf3 = (const float*)d_in[5];   // [10,150]
    float* out = (float*)d_out;                 // [10,2048]

    uint8_t* base = (uint8_t*)d_ws;
    size_t off = 0;
    auto carve = [&](size_t bytes) -> uint8_t* {
        uint8_t* p = base + off;
        off += (bytes + 255) & ~(size_t)255;
        return p;
    };
    _Float16* s1   = (_Float16*)carve(4704ull * T_BINS * 2);      // [n][t]
    _Float16* s2   = (_Float16*)carve((size_t)T_BINS * 1184 * 2); // [t][k]
    _Float16* s3   = (_Float16*)carve((size_t)T_BINS * 1600 * 2);
    _Float16* s4   = (_Float16*)carve((size_t)T_BINS * 320 * 2);
    _Float16* s5   = (_Float16*)carve((size_t)T_BINS * 160 * 2);
    float*    p3   = (float*)carve(16ull * 100 * T_BINS * 4);     // [16][100][t]
    float*    p4   = (float*)carve(304ull * T_BINS * 4);
    float*    p5   = (float*)carve(160ull * T_BINS * 4);
    float*    p6   = (float*)carve(16ull * T_BINS * 4);
    _Float16* Bim  = (_Float16*)carve(20ull * T_BINS * 160 * 2);  // im2col chunk
    _Float16* Wc2h = (_Float16*)carve(16ull * 160 * 2);
    _Float16* Wf1h = (_Float16*)carve(304ull * 1600 * 2);
    _Float16* Wf2h = (_Float16*)carve(160ull * 320 * 2);
    _Float16* Wf3h = (_Float16*)carve(16ull * 160 * 2);
    if (off > ws_size) return;

    auto gemm = [&](const _Float16* A, const _Float16* Bt, float* C,
                    int Mt, int Nt, int K, int ldc, int Mvalid) {
        int waves = Mt * Nt;              // always a multiple of 8 here
        int blocks = (waves + 7) / 8;     // 256 threads = 8 waves (wave32)
        wmma_gemm_f16<<<blocks, 256, 0, stream>>>(A, Bt, C, Mt, Nt, K, ldc, Mvalid);
    };

    // Weight conversion (f32 -> padded f16)
    pad_convert_f16<<<(16 * 160 + 255) / 256, 256, 0, stream>>>(Wc2, Wc2h, 16, 150, 16, 160);
    pad_convert_f16<<<(304 * 1600 + 255) / 256, 256, 0, stream>>>(Wf1, Wf1h, 300, 1600, 304, 1600);
    pad_convert_f16<<<(160 * 320 + 255) / 256, 256, 0, stream>>>(Wf2, Wf2h, 150, 300, 160, 320);
    pad_convert_f16<<<(16 * 160 + 255) / 256, 256, 0, stream>>>(Wf3, Wf3h, 10, 150, 16, 160);

    // Layer 1: conv1 (time-constant) + psp + spike
    conv1_psp_spike<<<(4704 + 127) / 128, 128, 0, stream>>>(x, Wc1, s1);
    // Layer 2: pool + psp + spike
    pool_psp_spike<<<(1176 + 127) / 128, 128, 0, stream>>>(s1, s2);

    // Layer 3: conv2 as implicit GEMM (M=16, K=160, N=100*T), 5 chunks of 20 pos
    for (int ch = 0; ch < 5; ++ch) {
        int chunkStart = ch * 20;
        im2col_conv2<<<(20 * 2048 * 160 + 255) / 256, 256, 0, stream>>>(s2, Bim, chunkStart);
        gemm(Wc2h, Bim, p3 + (size_t)chunkStart * T_BINS,
             /*Mt=*/1, /*Nt=*/(20 * T_BINS) / 16, /*K=*/160,
             /*ldc=*/100 * T_BINS, /*Mvalid=*/16);
    }
    psp_spike_r2c<<<(1600 + 127) / 128, 128, 0, stream>>>(p3, s3, 1600, 1600);

    // Layer 4: fc1  [300,1600] x [1600,2048]
    gemm(Wf1h, s3, p4, 19, T_BINS / 16, 1600, T_BINS, 304);
    psp_spike_r2c<<<(320 + 127) / 128, 128, 0, stream>>>(p4, s4, 300, 320);

    // Layer 5: fc2  [150,300] x [300,2048]
    gemm(Wf2h, s4, p5, 10, T_BINS / 16, 320, T_BINS, 160);
    psp_spike_r2c<<<(160 + 127) / 128, 128, 0, stream>>>(p5, s5, 150, 160);

    // Layer 6: fc3  [10,150] x [150,2048]
    gemm(Wf3h, s5, p6, 1, T_BINS / 16, 160, T_BINS, 16);
    psp_spike_final<<<1, 32, 0, stream>>>(p6, out);
}
